// EncodeProcessDecode_80487687127342
// MI455X (gfx1250) — compile-verified, matrix operand order checked by reading, not run on hardware
//
#include <hip/hip_runtime.h>
#include <hip/hip_bf16.h>

// ---------------------------------------------------------------------------
// EncodeProcessDecode GNN for gfx1250 (MI455X). bf16 WMMA GEMMs with fused
// gather (concat) on the A operand and fused bias/ReLU/LayerNorm/residual/
// segment-sum epilogues. Weights are DMA'd into LDS per block via the Tensor
// Data Mover (tensor_load_to_lds + s_wait_tensorcnt), read as WMMA B
// fragments with ds loads shared by all 4 waves. A fragments are software-
// pipelined one K-chunk ahead to hide gather latency behind the WMMA batch.
// ---------------------------------------------------------------------------

typedef __attribute__((ext_vector_type(16))) __bf16   v16bf;
typedef __attribute__((ext_vector_type(8)))  __bf16   v8bf;
typedef __attribute__((ext_vector_type(8)))  float    v8f;
typedef __attribute__((ext_vector_type(4)))  unsigned u32x4;
typedef __attribute__((ext_vector_type(8)))  unsigned u32x8;

#define LATENT   128
#define IN_SIZE  12
#define N_STEPS  8
#define LN_EPS   1e-5f
#define TPITCH   133   // epilogue tile pitch (odd -> conflict-free LDS banks)

// ---------------------------------------------------------------------------
// Small utility kernels
// ---------------------------------------------------------------------------
__global__ void k_zero_f32(float* p, long n) {
    long stride = (long)gridDim.x * blockDim.x;
    for (long i = (long)blockIdx.x * blockDim.x + threadIdx.x; i < n; i += stride)
        p[i] = 0.0f;
}

__global__ void k_zero_b16(__bf16* p, long n) {
    long stride = (long)gridDim.x * blockDim.x;
    for (long i = (long)blockIdx.x * blockDim.x + threadIdx.x; i < n; i += stride)
        p[i] = (__bf16)0.0f;
}

__global__ void k_f32_to_b16(const float* __restrict__ in, __bf16* __restrict__ out, long n) {
    long stride = (long)gridDim.x * blockDim.x;
    for (long i = (long)blockIdx.x * blockDim.x + threadIdx.x; i < n; i += stride)
        out[i] = (__bf16)in[i];
}

// Pack a row-major f32 weight matrix [K,128] into fragment-major bf16 so the
// GEMM reads each lane's 16-half B fragment as one contiguous 32B chunk.
// Element i: chunk=i>>9 (kc=chunk>>3, nt=chunk&7), lane=(i>>4)&31, e=i&15.
// B layout (A layout transposed): lane&15 = column in the 16-wide N tile;
// lane<16 holds K {0..7,16..23}, lane>=16 holds K {8..15,24..31}.
__global__ void k_pack_w(const float* __restrict__ W, __bf16* __restrict__ out, int K) {
    int i = blockIdx.x * 256 + threadIdx.x;
    int total = K * LATENT;
    if (i >= total) return;
    int e     = i & 15;
    int lane  = (i >> 4) & 31;
    int chunk = i >> 9;
    int kc = chunk >> 3;
    int nt = chunk & 7;
    int n  = nt * 16 + (lane & 15);
    int base0 = (lane < 16) ? 0 : 8;
    int k  = kc * 32 + ((e < 8) ? (base0 + e) : (base0 + 16 + (e - 8)));
    out[i] = (__bf16)W[k * LATENT + n];
}

// First encoder layer: [M,12] @ [12,128] + b, ReLU, write bf16. Scalar (tiny).
__global__ __launch_bounds__(128) void k_mlp_in12(const float* __restrict__ X,
                                                  const float* __restrict__ W,
                                                  const float* __restrict__ b,
                                                  __bf16* __restrict__ out, int M) {
    __shared__ float xr[IN_SIZE];
    int row = blockIdx.x;
    if (row >= M) return;
    if (threadIdx.x < IN_SIZE) xr[threadIdx.x] = X[(size_t)row * IN_SIZE + threadIdx.x];
    __syncthreads();
    int j = threadIdx.x;
    float a = b[j];
#pragma unroll
    for (int i = 0; i < IN_SIZE; ++i) a = fmaf(xr[i], W[i * LATENT + j], a);
    out[(size_t)row * LATENT + j] = (__bf16)fmaxf(a, 0.0f);
}

// Decoder MLP: 128 -> 3 -> 3 -> 3 (ReLU, ReLU, none). One thread per node.
__global__ void k_decoder(const float* __restrict__ xh,
                          const float* __restrict__ W1, const float* __restrict__ b1,
                          const float* __restrict__ W2, const float* __restrict__ b2,
                          const float* __restrict__ W3, const float* __restrict__ b3,
                          float* __restrict__ out, int M) {
    int r = blockIdx.x * 128 + threadIdx.x;
    if (r >= M) return;
    float h1[3] = {b1[0], b1[1], b1[2]};
    const float* xr = xh + (size_t)r * LATENT;
    for (int i = 0; i < LATENT; ++i) {
        float xv = xr[i];
#pragma unroll
        for (int j = 0; j < 3; ++j) h1[j] = fmaf(xv, W1[i * 3 + j], h1[j]);
    }
#pragma unroll
    for (int j = 0; j < 3; ++j) h1[j] = fmaxf(h1[j], 0.0f);
    float h2[3];
#pragma unroll
    for (int j = 0; j < 3; ++j) {
        float a = b2[j];
#pragma unroll
        for (int i = 0; i < 3; ++i) a = fmaf(h1[i], W2[i * 3 + j], a);
        h2[j] = fmaxf(a, 0.0f);
    }
#pragma unroll
    for (int j = 0; j < 3; ++j) {
        float a = b3[j];
#pragma unroll
        for (int i = 0; i < 3; ++i) a = fmaf(h2[i], W3[i * 3 + j], a);
        out[(size_t)r * 3 + j] = a;
    }
}

// ---------------------------------------------------------------------------
// Generic fused WMMA GEMM: C[M,128] = act( gatherConcat_A[M,K] @ W[K,128] + b )
// K is split into uniform 4-chunk (128-wide) regions, each with its own base
// pointer and optional row-index gather (implements [xh[dst]|xh[src]|eh] and
// [xh|aggr] concats without materializing). Weights: TDM DMA into LDS once
// per block; ds_load B fragments shared by all 4 waves. Epilogue: ReLU or
// LayerNorm + in-place f32 residual + bf16 mirror + atomicAdd segment-sum.
// ---------------------------------------------------------------------------
struct GemmDesc {
    const __bf16* A0; const int* idx0; int ld0;   // region 0: chunks 0..3
    const __bf16* A1; const int* idx1; int ld1;   // region 1: chunks 4..7
    const __bf16* A2; const int* idx2; int ld2;   // region 2: chunks 8..11
    int kChunks;           // K / 32  (4, 8 or 12)
    const __bf16* Wp;      // fragment-major packed weights (kChunks*8*512 halves)
    const float* bias;     // [128]
    const float* gamma;    // null => ReLU epilogue
    const float* beta;
    float* resid;          // f32 [M,128]; in-place residual or plain f32 out
    int residIsOut;        // 1: resid is output (encoder), 0: resid += LN(...)
    __bf16* outB;          // bf16 mirror / next-layer input [M,128]
    float* aggr;           // segment-sum target [Nnodes,128] or null
    const int* aggrIdx;    // dst index per row for scatter
    const __bf16* zeroRow; // zeroed bf16 row for out-of-range lanes
    int M;
};

__global__ __launch_bounds__(128) void k_gemm_bf16(GemmDesc g) {
    extern __shared__ char smem[];           // [0,wBytes): weights, then tile
    __bf16* wlds = (__bf16*)smem;
    const int wBytes = g.kChunks * 8192;     // kChunks * 8 tiles * 512 halves * 2B
    float* tile = (float*)(smem + wBytes);

    const int lane = threadIdx.x & 31;
    const int wave = threadIdx.x >> 5;

    // ---- TDM: DMA the packed weight block into LDS (wave 0 only). ----
    // D# per CDNA5 ISA: 1-row tile of kChunks*1024
    // 8-byte elements, contiguous; lds_addr = 0 (base of dynamic LDS segment).
    if (threadIdx.x < 32) {
        unsigned long long ga = (unsigned long long)(uintptr_t)g.Wp;
        unsigned units = (unsigned)(g.kChunks * 1024);   // # of 8B elements
        u32x4 d0;
        d0[0] = 1u;                                      // count=1, user mode
        d0[1] = 0u;                                      // lds_addr
        d0[2] = (unsigned)(ga & 0xFFFFFFFFu);            // global_addr[31:0]
        d0[3] = (unsigned)((ga >> 32) & 0x01FFFFFFu)     // global_addr[56:32]
              | (2u << 30);                              // type=2 ("image")
        u32x8 d1;
        d1[0] = 3u << 16;                                // data_size = 8B
        d1[1] = (units & 0xFFFFu) << 16;                 // tensor_dim0[15:0]
        d1[2] = ((units >> 16) & 0xFFFFu)                // tensor_dim0[31:16]
              | (1u << 16);                              // tensor_dim1 = 1
        d1[3] = (units & 0xFFFFu) << 16;                 // tile_dim0
        d1[4] = 1u;                                      // tile_dim1 = 1
        d1[5] = 0u; d1[6] = 0u; d1[7] = 0u;              // strides unused (1 row)
        asm volatile("tensor_load_to_lds %0, %1" :: "s"(d0), "s"(d1) : "memory");
        __builtin_amdgcn_s_wait_tensorcnt(0);
    }
    __syncthreads();

    const int rbase = blockIdx.x * 64 + wave * 16;
    const int myRow = rbase + (lane & 15);
    const bool rowOK = myRow < g.M;

    // Resolve per-lane row base pointers for each K region (gather fused).
    const __bf16* rp0 = g.zeroRow;
    const __bf16* rp1 = g.zeroRow;
    const __bf16* rp2 = g.zeroRow;
    if (rowOK) {
        long r0 = g.idx0 ? (long)g.idx0[myRow] : (long)myRow;
        rp0 = g.A0 + r0 * g.ld0;
        if (g.A1) { long r1 = g.idx1 ? (long)g.idx1[myRow] : (long)myRow; rp1 = g.A1 + r1 * g.ld1; }
        if (g.A2) { long r2 = g.idx2 ? (long)g.idx2[myRow] : (long)myRow; rp2 = g.A2 + r2 * g.ld2; }
    }

    v8f acc[8];
#pragma unroll
    for (int nt = 0; nt < 8; ++nt)
#pragma unroll
        for (int v = 0; v < 8; ++v) acc[nt][v] = 0.0f;

    const int laneK = (lane < 16) ? 0 : 8;   // per-lane K sub-offset (ISA A layout)

    // Prologue: A fragment for chunk 0 (two contiguous 16B loads per lane).
    v8bf lo = *(const v8bf*)(rp0 + laneK);
    v8bf hi = *(const v8bf*)(rp0 + laneK + 16);

    for (int kc = 0; kc < g.kChunks; ++kc) {
        // Software-pipelined prefetch of the next A fragment (branchless
        // region select: regions are uniform 4-chunk spans).
        int kn = (kc + 1 < g.kChunks) ? (kc + 1) : kc;
        const __bf16* np = (kn >= 8) ? rp2 : ((kn >= 4) ? rp1 : rp0);
        int nklo = (kn & 3) * 32 + laneK;
        v8bf nlo = *(const v8bf*)(np + nklo);
        v8bf nhi = *(const v8bf*)(np + nklo + 16);

        v16bf a;
#pragma unroll
        for (int i = 0; i < 8; ++i) { a[i] = lo[i]; a[i + 8] = hi[i]; }

        const __bf16* wb = wlds + (size_t)kc * 4096 + (size_t)lane * 16;
#pragma unroll
        for (int nt = 0; nt < 8; ++nt) {
            v16bf b = *(const v16bf*)(wb + (size_t)nt * 512);
            acc[nt] = __builtin_amdgcn_wmma_f32_16x16x32_bf16(
                false, a, false, b, (short)0, acc[nt], false, false);
        }
        lo = nlo; hi = nhi;
    }

    // Stage the wave's 16x128 f32 tile in LDS for the row-wise epilogue.
    float* my = tile + wave * (16 * TPITCH);
#pragma unroll
    for (int nt = 0; nt < 8; ++nt) {
#pragma unroll
        for (int v = 0; v < 8; ++v) {
            int r = ((lane >> 4) << 3) + v;   // lanes 0-15: rows 0-7, 16-31: 8-15
            int c = nt * 16 + (lane & 15);
            my[r * TPITCH + c] = acc[nt][v];
        }
    }
    __syncthreads();

    int t = threadIdx.x;
    if (t < 64) {
        int grow = blockIdx.x * 64 + t;
        if (grow < g.M) {
            float* rowp = tile + t * TPITCH;
            size_t obase = (size_t)grow * LATENT;
            if (g.gamma) {
                float mu = 0.0f;
                for (int c = 0; c < LATENT; ++c) {
                    float h = rowp[c] + g.bias[c];
                    rowp[c] = h;
                    mu += h;
                }
                mu *= (1.0f / LATENT);
                float var = 0.0f;
                for (int c = 0; c < LATENT; ++c) {
                    float d = rowp[c] - mu;
                    var = fmaf(d, d, var);
                }
                var *= (1.0f / LATENT);
                float rs = __frsqrt_rn(var + LN_EPS);
                size_t abase = 0;
                if (g.aggr) abase = (size_t)g.aggrIdx[grow] * LATENT;
                for (int c = 0; c < LATENT; ++c) {
                    float tv = (rowp[c] - mu) * rs * g.gamma[c] + g.beta[c];
                    float ov = tv;
                    if (g.resid && !g.residIsOut) ov = g.resid[obase + c] + tv;
                    if (g.resid) g.resid[obase + c] = ov;
                    if (g.outB) g.outB[obase + c] = (__bf16)ov;
                    if (g.aggr) atomicAdd(&g.aggr[abase + c], tv);  // segment_sum
                }
            } else {
                for (int c = 0; c < LATENT; ++c) {
                    float h = fmaxf(rowp[c] + g.bias[c], 0.0f);
                    g.outB[obase + c] = (__bf16)h;
                }
            }
        }
    }
}

// ---------------------------------------------------------------------------
// Host orchestration
// ---------------------------------------------------------------------------
static inline size_t alignup(size_t x) { return (x + 255) & ~(size_t)255; }

static void run_gemm(hipStream_t s, const GemmDesc& g) {
    int grid = (g.M + 63) / 64;
    size_t shmem = (size_t)g.kChunks * 8192 + (size_t)64 * TPITCH * 4;
    hipLaunchKernelGGL(k_gemm_bf16, dim3(grid), dim3(128), shmem, s, g);
}

extern "C" void kernel_launch(void* const* d_in, const int* in_sizes, int n_in,
                              void* d_out, int out_size, void* d_ws, size_t ws_size,
                              hipStream_t stream) {
    (void)n_in; (void)out_size; (void)ws_size;

    const int N = in_sizes[0] / IN_SIZE;        // nodes
    const int E = in_sizes[2] / 2;              // edges

    // Allow >64KB dynamic LDS for the GEMM kernel (gfx1250: 320KB/WGP).
    hipFuncSetAttribute((const void*)k_gemm_bf16,
                        hipFuncAttributeMaxDynamicSharedMemorySize,
                        12 * 8192 + 64 * TPITCH * 4);

    // --- input mapping (recursive dict-insertion order flatten) ---
    const float* x          = (const float*)d_in[0];
    const float* edge_attr  = (const float*)d_in[1];
    const int*   eidx       = (const int*)d_in[2];
    const int*   src        = eidx;             // edge_index[0]
    const int*   dst        = eidx + E;         // edge_index[1]

    // enc_node / enc_edge: W1,b1,W2,b2,W3,b3,g,beta
    const float* enW[2][3]; const float* enB[2][3]; const float* enG[2]; const float* enBeta[2];
    for (int m = 0; m < 2; ++m) {
        int base = 3 + m * 8;
        enW[m][0] = (const float*)d_in[base + 0]; enB[m][0] = (const float*)d_in[base + 1];
        enW[m][1] = (const float*)d_in[base + 2]; enB[m][1] = (const float*)d_in[base + 3];
        enW[m][2] = (const float*)d_in[base + 4]; enB[m][2] = (const float*)d_in[base + 5];
        enG[m]    = (const float*)d_in[base + 6]; enBeta[m] = (const float*)d_in[base + 7];
    }
    // steps.msg (19..26), steps.upd (27..34): W1,b1,W2,b2,W3,b3,g,beta ([8,...])
    const float* msgW1 = (const float*)d_in[19]; const float* msgB1 = (const float*)d_in[20];
    const float* msgW2 = (const float*)d_in[21]; const float* msgB2 = (const float*)d_in[22];
    const float* msgW3 = (const float*)d_in[23]; const float* msgB3 = (const float*)d_in[24];
    const float* msgG  = (const float*)d_in[25]; const float* msgBe = (const float*)d_in[26];
    const float* updW1 = (const float*)d_in[27]; const float* updB1 = (const float*)d_in[28];
    const float* updW2 = (const float*)d_in[29]; const float* updB2 = (const float*)d_in[30];
    const float* updW3 = (const float*)d_in[31]; const float* updB3 = (const float*)d_in[32];
    const float* updG  = (const float*)d_in[33]; const float* updBe = (const float*)d_in[34];
    // dec (35..40)
    const float* decW1 = (const float*)d_in[35]; const float* decB1 = (const float*)d_in[36];
    const float* decW2 = (const float*)d_in[37]; const float* decB2 = (const float*)d_in[38];
    const float* decW3 = (const float*)d_in[39]; const float* decB3 = (const float*)d_in[40];

    float* out = (float*)d_out;

    // --- workspace carve-up ---
    char* ws = (char*)d_ws;
    size_t off = 0;
    auto take = [&](size_t bytes) { char* p = ws + off; off += alignup(bytes); return p; };

    float*  xh_f32   = (float*) take((size_t)N * LATENT * 4);
    __bf16* xh_b16   = (__bf16*)take((size_t)N * LATENT * 2);
    float*  eh_f32   = (float*) take((size_t)E * LATENT * 4);
    __bf16* eh_b16   = (__bf16*)take((size_t)E * LATENT * 2);
    __bf16* h1       = (__bf16*)take((size_t)E * LATENT * 2);
    __bf16* h2       = (__bf16*)take((size_t)E * LATENT * 2);
    float*  aggr_f32 = (float*) take((size_t)N * LATENT * 4);
    __bf16* aggr_b16 = (__bf16*)take((size_t)N * LATENT * 2);
    __bf16* zeroRow  = (__bf16*)take(512 * 2);
    __bf16* encNodeW2p = (__bf16*)take((size_t)128 * 128 * 2);
    __bf16* encNodeW3p = (__bf16*)take((size_t)128 * 128 * 2);
    __bf16* encEdgeW2p = (__bf16*)take((size_t)128 * 128 * 2);
    __bf16* encEdgeW3p = (__bf16*)take((size_t)128 * 128 * 2);
    __bf16* msgW1p = (__bf16*)take((size_t)N_STEPS * 384 * 128 * 2);
    __bf16* msgW2p = (__bf16*)take((size_t)N_STEPS * 128 * 128 * 2);
    __bf16* msgW3p = (__bf16*)take((size_t)N_STEPS * 128 * 128 * 2);
    __bf16* updW1p = (__bf16*)take((size_t)N_STEPS * 256 * 128 * 2);
    __bf16* updW2p = (__bf16*)take((size_t)N_STEPS * 128 * 128 * 2);
    __bf16* updW3p = (__bf16*)take((size_t)N_STEPS * 128 * 128 * 2);

    auto packW = [&](const float* W, __bf16* dstp, int K) {
        int total = K * LATENT;
        hipLaunchKernelGGL(k_pack_w, dim3((total + 255) / 256), dim3(256), 0, stream, W, dstp, K);
    };

    // --- init & weight packing ---
    hipLaunchKernelGGL(k_zero_b16, dim3(1), dim3(256), 0, stream, zeroRow, (long)512);
    packW(enW[0][1], encNodeW2p, 128);
    packW(enW[0][2], encNodeW3p, 128);
    packW(enW[1][1], encEdgeW2p, 128);
    packW(enW[1][2], encEdgeW3p, 128);
    for (int s = 0; s < N_STEPS; ++s) {
        packW(msgW1 + (size_t)s * 384 * 128, msgW1p + (size_t)s * 384 * 128, 384);
        packW(msgW2 + (size_t)s * 128 * 128, msgW2p + (size_t)s * 128 * 128, 128);
        packW(msgW3 + (size_t)s * 128 * 128, msgW3p + (size_t)s * 128 * 128, 128);
        packW(updW1 + (size_t)s * 256 * 128, updW1p + (size_t)s * 256 * 128, 256);
        packW(updW2 + (size_t)s * 128 * 128, updW2p + (size_t)s * 128 * 128, 128);
        packW(updW3 + (size_t)s * 128 * 128, updW3p + (size_t)s * 128 * 128, 128);
    }

    GemmDesc g{};
    auto resetDesc = [&](GemmDesc& d) {
        d = GemmDesc{};
        d.zeroRow = zeroRow;
    };

    // --- encoders ---
    hipLaunchKernelGGL(k_mlp_in12, dim3(N), dim3(128), 0, stream, x, enW[0][0], enB[0][0], h1, N);
    resetDesc(g);
    g.A0 = h1; g.ld0 = LATENT; g.kChunks = 4;
    g.Wp = encNodeW2p; g.bias = enB[0][1]; g.outB = h2; g.M = N;
    run_gemm(stream, g);
    resetDesc(g);
    g.A0 = h2; g.ld0 = LATENT; g.kChunks = 4;
    g.Wp = encNodeW3p; g.bias = enB[0][2]; g.gamma = enG[0]; g.beta = enBeta[0];
    g.resid = xh_f32; g.residIsOut = 1; g.outB = xh_b16; g.M = N;
    run_gemm(stream, g);

    hipLaunchKernelGGL(k_mlp_in12, dim3(E), dim3(128), 0, stream, edge_attr, enW[1][0], enB[1][0], h1, E);
    resetDesc(g);
    g.A0 = h1; g.ld0 = LATENT; g.kChunks = 4;
    g.Wp = encEdgeW2p; g.bias = enB[1][1]; g.outB = h2; g.M = E;
    run_gemm(stream, g);
    resetDesc(g);
    g.A0 = h2; g.ld0 = LATENT; g.kChunks = 4;
    g.Wp = encEdgeW3p; g.bias = enB[1][2]; g.gamma = enG[1]; g.beta = enBeta[1];
    g.resid = eh_f32; g.residIsOut = 1; g.outB = eh_b16; g.M = E;
    run_gemm(stream, g);

    // --- processor: 8 message-passing steps ---
    for (int s = 0; s < N_STEPS; ++s) {
        hipLaunchKernelGGL(k_zero_f32, dim3(1024), dim3(256), 0, stream,
                           aggr_f32, (long)N * LATENT);

        // msg layer1: relu( [xh[dst] | xh[src] | eh] @ W1 + b1 )  (gather fused)
        resetDesc(g);
        g.A0 = xh_b16; g.idx0 = dst; g.ld0 = LATENT;
        g.A1 = xh_b16; g.idx1 = src; g.ld1 = LATENT;
        g.A2 = eh_b16; g.idx2 = nullptr; g.ld2 = LATENT;
        g.kChunks = 12;
        g.Wp = msgW1p + (size_t)s * 384 * 128;
        g.bias = msgB1 + s * LATENT; g.outB = h1; g.M = E;
        run_gemm(stream, g);

        // msg layer2: relu
        resetDesc(g);
        g.A0 = h1; g.ld0 = LATENT; g.kChunks = 4;
        g.Wp = msgW2p + (size_t)s * 128 * 128;
        g.bias = msgB2 + s * LATENT; g.outB = h2; g.M = E;
        run_gemm(stream, g);

        // msg layer3: LN, eh += new_e (in place), scatter segment_sum into aggr
        resetDesc(g);
        g.A0 = h2; g.ld0 = LATENT; g.kChunks = 4;
        g.Wp = msgW3p + (size_t)s * 128 * 128;
        g.bias = msgB3 + s * LATENT; g.gamma = msgG + s * LATENT; g.beta = msgBe + s * LATENT;
        g.resid = eh_f32; g.residIsOut = 0; g.outB = eh_b16;
        g.aggr = aggr_f32; g.aggrIdx = dst; g.M = E;
        run_gemm(stream, g);

        hipLaunchKernelGGL(k_f32_to_b16, dim3(1024), dim3(256), 0, stream,
                           aggr_f32, aggr_b16, (long)N * LATENT);

        // upd layer1: relu( [xh | aggr] @ W1 + b1 )
        resetDesc(g);
        g.A0 = xh_b16;   g.ld0 = LATENT;
        g.A1 = aggr_b16; g.ld1 = LATENT;
        g.kChunks = 8;
        g.Wp = updW1p + (size_t)s * 256 * 128;
        g.bias = updB1 + s * LATENT; g.outB = h1; g.M = N;
        run_gemm(stream, g);

        // upd layer2: relu
        resetDesc(g);
        g.A0 = h1; g.ld0 = LATENT; g.kChunks = 4;
        g.Wp = updW2p + (size_t)s * 128 * 128;
        g.bias = updB2 + s * LATENT; g.outB = h2; g.M = N;
        run_gemm(stream, g);

        // upd layer3: LN, xh += new_x (in place)
        resetDesc(g);
        g.A0 = h2; g.ld0 = LATENT; g.kChunks = 4;
        g.Wp = updW3p + (size_t)s * 128 * 128;
        g.bias = updB3 + s * LATENT; g.gamma = updG + s * LATENT; g.beta = updBe + s * LATENT;
        g.resid = xh_f32; g.residIsOut = 0; g.outB = xh_b16; g.M = N;
        run_gemm(stream, g);
    }

    // --- decoder ---
    hipLaunchKernelGGL(k_decoder, dim3((N + 127) / 128), dim3(128), 0, stream,
                       xh_f32, decW1, decB1, decW2, decB2, decW3, decB3, out, N);
}
